// causal_self_attention_77283641524904
// MI455X (gfx1250) — matrix operand order repaired and checked by reading, not compiled, not run on hardware
//
#include <hip/hip_runtime.h>

// ---------------------------------------------------------------------------
// Causal self-attention, B=4 L=2048 C=1024, single head.
// [k_cvt x3] -> [k_gemm<QKV> (TDM-staged)] -> [k_scores] -> [k_softmax]
//            -> [k_pv] -> [k_gemm<OUT> (TDM-staged)]
// GEMMs: v_wmma_f32_16x16x32_bf16; projections stage tiles in LDS via
// tensor_load_to_lds (TENSORcnt, double-buffered).
// ---------------------------------------------------------------------------

typedef __attribute__((ext_vector_type(16))) __bf16  v16bf;
typedef __attribute__((ext_vector_type(8)))  __bf16  bf16x8;
typedef __attribute__((ext_vector_type(4)))  __bf16  bf16x4;
typedef __attribute__((ext_vector_type(8)))  float   v8f;
typedef __attribute__((ext_vector_type(4)))  float   f32x4;
typedef __attribute__((ext_vector_type(4)))  unsigned int u32x4;
typedef __attribute__((ext_vector_type(4)))  int     i32x4;
typedef __attribute__((ext_vector_type(8)))  int     i32x8;

#define WMMA_BF16(a, b, c) \
  __builtin_amdgcn_wmma_f32_16x16x32_bf16(false, (a), false, (b), (short)0, (c), false, false)

static constexpr int Bb = 4;
static constexpr int Ll = 2048;
static constexpr int Cc = 1024;

// ---------------------------------------------------------------------------
// TDM: DMA a 2D tile (tileRows x 32 bf16, row stride = strideElems) from
// global into LDS at lds_addr (D# per CDNA5 ISA).
// ---------------------------------------------------------------------------
__device__ __forceinline__ void tdm_load_tile(unsigned lds_addr, const void* gptr,
                                              unsigned tileRows, unsigned tensorK,
                                              unsigned tensorRows,
                                              unsigned long long strideElems) {
  const unsigned long long ga = (unsigned long long)(uintptr_t)gptr;
  u32x4 g0;
  g0[0] = 1u;                                                // count=1 (user D#)
  g0[1] = lds_addr;                                          // bits 63:32
  g0[2] = (unsigned)(ga & 0xffffffffu);                      // global_addr[31:0]
  g0[3] = (unsigned)((ga >> 32) & 0x01ffffffu) | (2u << 30); // addr[56:32] | type=2
  i32x8 g1;
  g1[0] = (int)(1u << 16);                                   // data_size=1 (2B)
  g1[1] = (int)((tensorK & 0xffffu) << 16);                  // tensor_dim0[15:0]
  g1[2] = (int)(((tensorK >> 16) & 0xffffu) | ((tensorRows & 0xffffu) << 16));
  g1[3] = (int)(((tensorRows >> 16) & 0xffffu) | (32u << 16)); // tile_dim0=32
  g1[4] = (int)(tileRows & 0xffffu);                         // tile_dim1 (dim2=0)
  g1[5] = (int)(strideElems & 0xffffffffu);                  // dim0_stride[31:0]
  g1[6] = (int)((strideElems >> 32) & 0xffffu);              // stride[47:32]
  g1[7] = 0;
  i32x4 z4 = {0, 0, 0, 0};
#if __clang_major__ >= 23
  i32x8 z8 = {0, 0, 0, 0, 0, 0, 0, 0};
  __builtin_amdgcn_tensor_load_to_lds(g0, g1, z4, z4, z8, 0);
#else
  __builtin_amdgcn_tensor_load_to_lds(g0, g1, z4, z4, 0);
#endif
}

// ---- A fragment (16x32 bf16): lane = row, 16 contiguous K at ks+h*16.
__device__ __forceinline__ v16bf a_frag_bf16(const __bf16* __restrict__ A, int lda,
                                             int m0, int ks, int lane) {
  const int hf = (lane >> 4) & 1, mr = lane & 15;
  const __bf16* p = A + (size_t)(m0 + mr) * lda + ks + hf * 16;
  bf16x8 lo = *(const bf16x8*)p;
  bf16x8 hi = *(const bf16x8*)(p + 8);
  v16bf a;
#pragma unroll
  for (int i = 0; i < 8; ++i) { a[i] = lo[i]; a[8 + i] = hi[i]; }
  return a;
}

// ---- B fragment (32x16 bf16): lane = column, 16 contiguous K at ks+h*16.
__device__ __forceinline__ v16bf b_frag_bf16(const __bf16* __restrict__ Bsrc, int ldb,
                                             int n0, int ks, int lane) {
  const int hf = (lane >> 4) & 1, nr = lane & 15;
  const __bf16* p = Bsrc + (size_t)(n0 + nr) * ldb + ks + hf * 16;
  return *(const v16bf*)p;
}

// ---------------------------------------------------------------------------
// Elementwise f32 -> bf16 (x4 vectors). One-shot, ~25 MB written total.
// ---------------------------------------------------------------------------
__global__ void k_cvt(const float* __restrict__ s, __bf16* __restrict__ d, int n4) {
  const int i = blockIdx.x * 256 + threadIdx.x;
  if (i >= n4) return;
  f32x4 v = ((const f32x4*)s)[i];
  bf16x4 o;
#pragma unroll
  for (int j = 0; j < 4; ++j) o[j] = (__bf16)v[j];
  ((bf16x4*)d)[i] = o;
}

// ---------------------------------------------------------------------------
// TDM-staged block GEMM: C(MxN) = A(MxK,bf16) * B(NxK,bf16)^T.
// Block tile 128x128, 8 waves, wave tile 32x64 (8 WMMAs / K-step).
// Wave 0 DMAs A(128x32) + B(128x32) into double-buffered LDS.
// MODE 0: QKV epilogue (Q scaled 1/32, K row-major, V transposed, bf16).
// MODE 1: plain f32 store.
// ---------------------------------------------------------------------------
template <int MODE>
__global__ __launch_bounds__(256) void k_gemm(const __bf16* __restrict__ Asrc,
                                              const __bf16* __restrict__ Bsrc,
                                              int Kdim, int Mtot, int Ntot,
                                              void* __restrict__ out0,
                                              void* __restrict__ out1,
                                              void* __restrict__ out2) {
  __shared__ __bf16 Abuf[2][128][32];
  __shared__ __bf16 Bbuf[2][128][32];

  const int lane = threadIdx.x & 31;
  const int wave = __builtin_amdgcn_readfirstlane(threadIdx.x >> 5); // scalar
  const int mi = wave >> 1;               // 0..3: 32-row strip
  const int ni = wave & 1;                // 0..1: 64-col strip
  const int m0b = blockIdx.y * 128;
  const int n0b = blockIdx.x * 128;
  const int nsteps = Kdim / 32;

  if (wave == 0) {                        // scalar branch: stage K-step 0
    tdm_load_tile((unsigned)(uintptr_t)&Abuf[0][0][0], Asrc + (size_t)m0b * Kdim,
                  128, (unsigned)Kdim, (unsigned)Mtot, (unsigned)Kdim);
    tdm_load_tile((unsigned)(uintptr_t)&Bbuf[0][0][0], Bsrc + (size_t)n0b * Kdim,
                  128, (unsigned)Kdim, (unsigned)Ntot, (unsigned)Kdim);
  }

  v8f acc[2][4] = {};
  const int hf = (lane >> 4) & 1, lr = lane & 15;

  for (int s = 0; s < nsteps; ++s) {
    const int cur = s & 1;
    if (wave == 0) {
      if (s + 1 < nsteps) {               // prefetch next tiles into other buffer
        const int ks = (s + 1) * 32;
        tdm_load_tile((unsigned)(uintptr_t)&Abuf[cur ^ 1][0][0],
                      Asrc + (size_t)m0b * Kdim + ks, 128, (unsigned)Kdim,
                      (unsigned)Mtot, (unsigned)Kdim);
        tdm_load_tile((unsigned)(uintptr_t)&Bbuf[cur ^ 1][0][0],
                      Bsrc + (size_t)n0b * Kdim + ks, 128, (unsigned)Kdim,
                      (unsigned)Ntot, (unsigned)Kdim);
        __builtin_amdgcn_s_wait_tensorcnt(2);  // current pair done (in-order)
      } else {
        __builtin_amdgcn_s_wait_tensorcnt(0);
      }
    }
    __syncthreads();                      // tiles visible to all waves

    v16bf af[2];
#pragma unroll
    for (int t = 0; t < 2; ++t) {
      const __bf16* p = &Abuf[cur][mi * 32 + t * 16 + lr][hf * 8];
      bf16x8 lo = *(const bf16x8*)p;
      bf16x8 hi = *(const bf16x8*)(p + 16);
#pragma unroll
      for (int i = 0; i < 8; ++i) { af[t][i] = lo[i]; af[t][8 + i] = hi[i]; }
    }
#pragma unroll
    for (int j = 0; j < 4; ++j) {
      const __bf16* p = &Bbuf[cur][ni * 64 + j * 16 + lr][hf * 8];
      bf16x8 blo = *(const bf16x8*)p;
      bf16x8 bhi = *(const bf16x8*)(p + 16);
      v16bf bf;
#pragma unroll
      for (int i = 0; i < 8; ++i) { bf[i] = blo[i]; bf[8 + i] = bhi[i]; }
      acc[0][j] = WMMA_BF16(af[0], bf, acc[0][j]);
      acc[1][j] = WMMA_BF16(af[1], bf, acc[1][j]);
    }
    __syncthreads();                      // reads done before next DMA overwrite
  }

#pragma unroll
  for (int t = 0; t < 2; ++t)
#pragma unroll
    for (int j = 0; j < 4; ++j) {
      const int n = n0b + ni * 64 + j * 16 + lr;
#pragma unroll
      for (int r = 0; r < 8; ++r) {
        const int m = m0b + mi * 32 + t * 16 + r + hf * 8;
        const float v = acc[t][j][r];
        if (MODE == 0) {                  // QKV split (seg uniform per tile)
          const int seg = n >> 10, c = n & 1023;
          if (seg == 0)      ((__bf16*)out0)[(size_t)m * Cc + c] = (__bf16)(v * 0.03125f);
          else if (seg == 1) ((__bf16*)out1)[(size_t)m * Cc + c] = (__bf16)v;
          else {
            const int bb = m >> 11, l = m & 2047;
            ((__bf16*)out2)[((size_t)bb * Cc + c) * Ll + l] = (__bf16)v; // V^T
          }
        } else {
          ((float*)out0)[(size_t)m * Ntot + n] = v;
        }
      }
    }
}

// ---------------------------------------------------------------------------
// S = Q*K^T (Q pre-scaled). Strips of 4 ktiles per wave; scalar control flow.
// Compute all 4 tiles unconditionally (constant acc indices -> no movrel;
// over-computed tiles only occur on the single diagonal strip per qtile and
// read rows that always exist). Store guarded per-tile by scalar compare;
// elementwise causal mask only fires on the diagonal tile.
// ---------------------------------------------------------------------------
__global__ void k_scores(const __bf16* __restrict__ Qb, const __bf16* __restrict__ Kb,
                         __bf16* __restrict__ S) {
  const int lane = threadIdx.x & 31;
  const int wave = __builtin_amdgcn_readfirstlane(threadIdx.x >> 5); // scalar
  const int qt = blockIdx.y;
  const int kt0 = (blockIdx.x * 8 + wave) * 4;
  if (kt0 > qt) return;                       // scalar exit
  const int b = blockIdx.z;
  const __bf16* Q = Qb + (size_t)b * Ll * Cc;
  const __bf16* K = Kb + (size_t)b * Ll * Cc;
  __bf16* Sb = S + (size_t)b * Ll * Ll;
  const int m0 = qt * 16;

  v8f acc[4] = {};
  for (int ks = 0; ks < Cc; ks += 32) {
    v16bf a = a_frag_bf16(Q, Cc, m0, ks, lane);
#pragma unroll
    for (int j = 0; j < 4; ++j) {             // guard-free, constant indices
      v16bf bf = b_frag_bf16(K, Cc, (kt0 + j) * 16, ks, lane);
      acc[j] = WMMA_BF16(a, bf, acc[j]);
    }
  }

  const int hf = (lane >> 4) & 1, nr = lane & 15;
#pragma unroll
  for (int j = 0; j < 4; ++j) {
    const int kt = kt0 + j;
    if (kt <= qt) {                           // scalar guard, unrolled j
#pragma unroll
      for (int r = 0; r < 8; ++r) {
        const int q = m0 + r + hf * 8;
        const int k = kt * 16 + nr;
        float v = acc[j][r];
        if (k > q) v = -__builtin_inff();     // only fires on diagonal tile
        Sb[(size_t)q * Ll + k] = (__bf16)v;
      }
    }
  }
}

// ---------------------------------------------------------------------------
// Row softmax (in place), causal length (qtile+1)*16, zero-pad to mult of 32.
// Fixed 8-slot register tile (constant indices; OOB slots = -inf -> exp = 0).
// ---------------------------------------------------------------------------
__global__ void k_softmax(__bf16* __restrict__ S) {
  const int q = blockIdx.x, b = blockIdx.y, tid = threadIdx.x;
  __bf16* row = S + ((size_t)b * Ll + q) * Ll;
  const int len = ((q >> 4) + 1) * 16;
  const int len32 = (len + 31) & ~31;

  float vals[8];
  float m = -__builtin_inff();
#pragma unroll
  for (int jj = 0; jj < 8; ++jj) {
    const int i = tid + jj * 256;
    const float v = (i < len) ? (float)row[i] : -__builtin_inff();
    vals[jj] = v;
    m = fmaxf(m, v);
  }

  __shared__ float red[256];
  red[tid] = m;
  __syncthreads();
  for (int s = 128; s > 0; s >>= 1) { if (tid < s) red[tid] = fmaxf(red[tid], red[tid + s]); __syncthreads(); }
  const float rowmax = red[0];                // finite: diagonal always present
  __syncthreads();

  float sum = 0.f;
#pragma unroll
  for (int jj = 0; jj < 8; ++jj) {            // exp(-inf - rowmax) == 0
    vals[jj] = __expf(vals[jj] - rowmax);
    sum += vals[jj];
  }
  red[tid] = sum;
  __syncthreads();
  for (int s = 128; s > 0; s >>= 1) { if (tid < s) red[tid] += red[tid + s]; __syncthreads(); }
  const float inv = 1.0f / red[0];

#pragma unroll
  for (int jj = 0; jj < 8; ++jj) {
    const int i = tid + jj * 256;
    if (i < len) row[i] = (__bf16)(vals[jj] * inv);
  }
  for (int i = len + tid; i < len32; i += 256) row[i] = (__bf16)0.0f; // pad
}

// ---------------------------------------------------------------------------
// O = P * V (V transposed => contiguous-K B fragments). Scalar K-loop bound.
// ---------------------------------------------------------------------------
__global__ void k_pv(const __bf16* __restrict__ P, const __bf16* __restrict__ Vt,
                     __bf16* __restrict__ O) {
  const int lane = threadIdx.x & 31;
  const int wave = __builtin_amdgcn_readfirstlane(threadIdx.x >> 5); // scalar
  const int n0 = blockIdx.x * 64;
  const int mt = blockIdx.y * 8 + wave;
  const int b = mt >> 7, qt = mt & 127;
  const __bf16* Pb = P + (size_t)b * Ll * Ll;
  const __bf16* Vb = Vt + (size_t)b * Cc * Ll;
  const int m0 = qt * 16;
  const int len32 = (((qt + 1) * 16) + 31) & ~31;

  v8f acc[4] = {};
  for (int ks = 0; ks < len32; ks += 32) {    // scalar trip count
    v16bf a = a_frag_bf16(Pb, Ll, m0, ks, lane);
#pragma unroll
    for (int j = 0; j < 4; ++j) {
      v16bf bf = b_frag_bf16(Vb, Ll, n0 + j * 16, ks, lane);
      acc[j] = WMMA_BF16(a, bf, acc[j]);
    }
  }

  const int hf = (lane >> 4) & 1, nr = lane & 15;
#pragma unroll
  for (int j = 0; j < 4; ++j)
#pragma unroll
    for (int r = 0; r < 8; ++r) {
      const int q = m0 + r + hf * 8;
      const int c = n0 + j * 16 + nr;
      O[((size_t)b * Ll + q) * Cc + c] = (__bf16)acc[j][r];
    }
}

// ---------------------------------------------------------------------------
extern "C" void kernel_launch(void* const* d_in, const int* in_sizes, int n_in,
                              void* d_out, int out_size, void* d_ws, size_t ws_size,
                              hipStream_t stream) {
  const float* net_in = (const float*)d_in[0]; // (4, 2048, 1024)
  const float* w_qkv  = (const float*)d_in[1]; // (3072, 1024)
  const float* w_out  = (const float*)d_in[2]; // (1024, 1024)
  float* y = (float*)d_out;

  // ws (bf16): Xb | Wqkvb | Woutb | Q | K | V^T | S/P | O   (~126 MiB)
  char* ws = (char*)d_ws;
  const size_t nX = (size_t)Bb * Ll * Cc;      // 8.39M
  const size_t nWq = (size_t)3 * Cc * Cc;      // 3.15M
  const size_t nWo = (size_t)Cc * Cc;          // 1.05M
  const size_t nS = (size_t)Bb * Ll * Ll;      // 16.8M
  __bf16* Xb    = (__bf16*)ws;                          ws += nX * 2;
  __bf16* Wqkvb = (__bf16*)ws;                          ws += nWq * 2;
  __bf16* Woutb = (__bf16*)ws;                          ws += nWo * 2;
  __bf16* Qb    = (__bf16*)ws;                          ws += nX * 2;
  __bf16* Kb    = (__bf16*)ws;                          ws += nX * 2;
  __bf16* Vt    = (__bf16*)ws;                          ws += nX * 2;
  __bf16* S     = (__bf16*)ws;                          ws += nS * 2;
  __bf16* O     = (__bf16*)ws;

  k_cvt<<<dim3((unsigned)((nX / 4 + 255) / 256)), 256, 0, stream>>>(net_in, Xb, (int)(nX / 4));
  k_cvt<<<dim3((unsigned)((nWq / 4 + 255) / 256)), 256, 0, stream>>>(w_qkv, Wqkvb, (int)(nWq / 4));
  k_cvt<<<dim3((unsigned)((nWo / 4 + 255) / 256)), 256, 0, stream>>>(w_out, Woutb, (int)(nWo / 4));

  // QKV projection: M=8192, N=3072, K=1024
  k_gemm<0><<<dim3(24, 64, 1), 256, 0, stream>>>(Xb, Wqkvb, Cc, Bb * Ll, 3 * Cc,
                                                 Qb, Kb, Vt);
  k_scores <<<dim3(4, 128, 4),  256, 0, stream>>>(Qb, Kb, S);
  k_softmax<<<dim3(2048, 4, 1), 256, 0, stream>>>(S);
  k_pv     <<<dim3(16, 64, 1),  256, 0, stream>>>(S, Vt, O);
  // Output projection: M=8192, N=1024, K=1024
  k_gemm<1><<<dim3(8, 64, 1),  256, 0, stream>>>(O, Woutb, Cc, Bb * Ll, Cc,
                                                 y, nullptr, nullptr);
}
